// CrossModalAttention_78752520339838
// MI455X (gfx1250) — matrix operand order repaired and checked by reading, compiled, not run on hardware
//
#include <hip/hip_runtime.h>
#include <hip/hip_bf16.h>

// ---------------------------------------------------------------------------
// CrossModalAttention on MI455X (gfx1250): bf16 WMMA pipeline, v2.1.
//   B=4, H=8, L=2048, D=64, C=512
//   K0a: transpose ppg/ecg (B,C,L)f32 -> (B,L,C)bf16 via LDS tile
//        (async global->LDS copy where the toolchain exposes it)
//   K0b: convert Wq/Wk/Wv/Wo to bf16 once (v_perm_b32 pair packing)
//   K1 : QKV projections, 16x64 tile per wave, pure-bf16 WMMA loop
//        (A frag = 2x global_load_b128, B frag = 2x global_load_b128)
//   K2 : attention per (b,h,16 rows): scores in LDS (16x2048 f32 fits in
//        320KB LDS), softmax in LDS, P@V via WMMA (A from ds_load_b64+v_perm)
//   K3 : ctx @ Wo^T + bo + residual, stored transposed (B,C,L)
// All matmuls: v_wmma_f32_16x16x32_bf16 (D = A*B + C, f32 accumulate).
// ---------------------------------------------------------------------------

typedef __bf16 bf16_t;
typedef __attribute__((ext_vector_type(16))) __bf16    v16bf;
typedef __attribute__((ext_vector_type(8)))  float     v8f;
typedef __attribute__((ext_vector_type(4)))  float     v4f;
typedef __attribute__((ext_vector_type(4)))  int       v4i;
typedef __attribute__((ext_vector_type(4)))  unsigned  v4u;
typedef __attribute__((ext_vector_type(8)))  unsigned  v8u;

#define NB 4
#define NH 8
#define NL 2048
#define ND 64
#define NC 512
#define SSTR 2052  // padded LDS row stride for the 16 x 2048 score panel

#if defined(__HIP_DEVICE_COMPILE__) && \
    __has_builtin(__builtin_amdgcn_global_load_async_to_lds_b128) && \
    __has_builtin(__builtin_amdgcn_s_wait_asynccnt)
#define USE_ASYNC 1
#else
#define USE_ASYNC 0
#endif

// pack two f32 into two bf16 (truncation) with a single v_perm_b32
__device__ __forceinline__ unsigned pack_bf16(float lo, float hi) {
  return __builtin_amdgcn_perm(__builtin_bit_cast(unsigned, hi),
                               __builtin_bit_cast(unsigned, lo),
                               0x07060302u);
}

__device__ __forceinline__ bf16_t f2bf(float f) {
  unsigned u = __builtin_bit_cast(unsigned, f);
  unsigned short h = (unsigned short)(u >> 16);   // truncate
  return __builtin_bit_cast(bf16_t, h);
}

// A-fragment (16-bit, 16x32), contiguous k order matching the B loads:
// element s of lane half khalf holds k = khalf*16 + s, i.e. dwords
//   khalf*8 .. +3   (k = khalf*16 .. +7)
//   khalf*8+4 .. +7 (k = khalf*16+8 .. +15)
// -> A fragment = two global_load_b128 from a row-major bf16 row.
__device__ __forceinline__ v16bf load_a_frag(const bf16_t* p32, int khalf) {
  const v4u* p = (const v4u*)p32;
  const v4u lo = p[2 * khalf];
  const v4u hi = p[2 * khalf + 1];
  v8u r;
  r[0] = lo[0]; r[1] = lo[1]; r[2] = lo[2]; r[3] = lo[3];
  r[4] = hi[0]; r[5] = hi[1]; r[6] = hi[2]; r[7] = hi[3];
  return __builtin_bit_cast(v16bf, r);
}

// ---------------------------------------------------------------------------
// K0a: transpose (B,C,L) f32 -> (B,L,C) bf16.  grid=(L/32, C/32, 2*B), 256 thr.
// ---------------------------------------------------------------------------
__global__ __launch_bounds__(256) void transpose_kernel(
    const float* __restrict__ ppg, const float* __restrict__ ecg,
    bf16_t* __restrict__ xbf, bf16_t* __restrict__ ybf)
{
  __shared__ float tile[32 * 36];     // [c][l], stride 36 keeps 16B alignment
  const int tid = threadIdx.x;
  const int l0  = blockIdx.x * 32;
  const int c0  = blockIdx.y * 32;
  const int b   = blockIdx.z & 3;
  const int sel = blockIdx.z >> 2;
  const float* src = ((sel == 0) ? ppg : ecg) + (size_t)b * NC * NL;
  bf16_t*      dst = ((sel == 0) ? xbf : ybf) + (size_t)b * NL * NC;

  // stage a 32(c) x 32(l) f32 tile: 16B per lane, coalesced along l
  const int row = tid >> 3;   // c within tile
  const int seg = tid & 7;    // 16B segment
  const float* g = src + (size_t)(c0 + row) * NL + l0 + seg * 4;
#if USE_ASYNC
  __builtin_amdgcn_global_load_async_to_lds_b128(
      (__attribute__((address_space(1))) v4i*)g,
      (__attribute__((address_space(3))) v4i*)&tile[row * 36 + seg * 4],
      0, 0);
  __builtin_amdgcn_s_wait_asynccnt(0);
#else
  *(v4f*)&tile[row * 36 + seg * 4] = *(const v4f*)g;
#endif
  __syncthreads();

  // write transposed, packing bf16 pairs along c (coalesced b32 stores)
  const int cx = (tid & 15) * 2;
  const int ly = tid >> 4;
#pragma unroll
  for (int r = 0; r < 2; ++r) {
    const int l = ly + r * 16;
    const unsigned pk =
        pack_bf16(tile[cx * 36 + l], tile[(cx + 1) * 36 + l]);
    *(unsigned*)&dst[(size_t)(l0 + l) * NC + c0 + cx] = pk;
  }
}

// ---------------------------------------------------------------------------
// K0b: f32 -> bf16 weight conversion (all four matrices). 524288 pairs.
// ---------------------------------------------------------------------------
__global__ __launch_bounds__(256) void wconv_kernel(
    const float* __restrict__ Wq, const float* __restrict__ Wk,
    const float* __restrict__ Wv, const float* __restrict__ Wo,
    bf16_t* __restrict__ wqb, bf16_t* __restrict__ wkb,
    bf16_t* __restrict__ wvb, bf16_t* __restrict__ wob)
{
  const int g = blockIdx.x * 256 + threadIdx.x;  // pair index
  const int w = g >> 17;                          // 131072 pairs per matrix
  const int off = (g & 131071) * 2;
  const float* W = (w == 0) ? Wq : (w == 1) ? Wk : (w == 2) ? Wv : Wo;
  bf16_t*     Wb = (w == 0) ? wqb : (w == 1) ? wkb : (w == 2) ? wvb : wob;
  const float2 v = *(const float2*)(W + off);
  *(unsigned*)&Wb[off] = pack_bf16(v.x, v.y);
}

// ---------------------------------------------------------------------------
// K1: QKV projections. grid=(L/16, C/64, 3*B), 1 wave; 16x64 tile, 64 WMMAs.
// sel: 0 -> Q (from xbf), 1 -> K (from ybf), 2 -> V transposed (from ybf)
// ---------------------------------------------------------------------------
__global__ __launch_bounds__(32) void qkv_kernel(
    const bf16_t* __restrict__ xbf, const bf16_t* __restrict__ ybf,
    const bf16_t* __restrict__ wqb, const float* __restrict__ bq,
    const bf16_t* __restrict__ wkb, const float* __restrict__ bk,
    const bf16_t* __restrict__ wvb, const float* __restrict__ bv,
    bf16_t* __restrict__ qbuf, bf16_t* __restrict__ kbuf,
    bf16_t* __restrict__ vtbuf)
{
  const int lid   = threadIdx.x & 31;
  const int ln    = lid & 15;
  const int khalf = lid >> 4;
  const int l0    = blockIdx.x * 16;
  const int h     = blockIdx.y;          // c0 = h*64 (one head per block)
  const int c0    = h * 64;
  const int b     = blockIdx.z & 3;
  const int sel   = blockIdx.z >> 2;

  const bf16_t* X    = ((sel == 0) ? xbf : ybf) + (size_t)b * NL * NC;
  const bf16_t* W    = (sel == 0) ? wqb : (sel == 1) ? wkb : wvb;
  const float*  bias = (sel == 0) ? bq : (sel == 1) ? bk : bv;

  const bf16_t* arow = X + (size_t)(l0 + ln) * NC;
  v8f acc[4] = {{}, {}, {}, {}};
#pragma unroll
  for (int kb = 0; kb < 16; ++kb) {
    const int kbase = kb * 32;
    const v16bf a = load_a_frag(arow + kbase, khalf);
#pragma unroll
    for (int nt = 0; nt < 4; ++nt) {
      const bf16_t* wrow =
          W + (size_t)(c0 + nt * 16 + ln) * NC + kbase + khalf * 16;
      const v16bf bm = *(const v16bf*)wrow;
      acc[nt] = __builtin_amdgcn_wmma_f32_16x16x32_bf16(
          false, a, false, bm, (short)0, acc[nt], false, false);
    }
  }

#pragma unroll
  for (int nt = 0; nt < 4; ++nt) {
    const int c = c0 + nt * 16 + ln;
    const int d = c & 63;
    const float bb = bias[c];
#pragma unroll
    for (int r = 0; r < 8; ++r) {
      const int l = l0 + khalf * 8 + r;
      const float v = acc[nt][r] + bb;
      if (sel == 2) {
        vtbuf[((size_t)(b * NH + h) * ND + d) * NL + l] = f2bf(v);
      } else {
        bf16_t* dst = (sel == 0) ? qbuf : kbuf;
        dst[((size_t)(b * NH + h) * NL + l) * ND + d] = f2bf(v);
      }
    }
  }
}

// ---------------------------------------------------------------------------
// K2: attention for one (b,h) and 16 query rows. grid=(L/16, B*H), 128 thr.
// Dynamic LDS: scores[16][SSTR] f32 | red[16][8] | ctxRed[4][16][64]
// ---------------------------------------------------------------------------
__global__ __launch_bounds__(128) void attn_kernel(
    const bf16_t* __restrict__ qbuf, const bf16_t* __restrict__ kbuf,
    const bf16_t* __restrict__ vtbuf, bf16_t* __restrict__ ctxbuf)
{
  extern __shared__ float smem[];
  float* scores = smem;                  // 16 * SSTR
  float* red    = scores + 16 * SSTR;    // 16 * 8
  float* ctxRed = red + 16 * 8;          // 4 * 16 * 64

  const int tid   = threadIdx.x;
  const int wave  = tid >> 5;
  const int lid   = tid & 31;
  const int ln    = lid & 15;
  const int khalf = lid >> 4;

  const int q0 = blockIdx.x * 16;
  const int bh = blockIdx.y;             // b*NH + h

  const bf16_t* Q  = qbuf  + (size_t)bh * NL * ND;
  const bf16_t* K  = kbuf  + (size_t)bh * NL * ND;
  const bf16_t* Vt = vtbuf + (size_t)bh * ND * NL;

  // Q A-fragments (k 0..31, 32..63), two b128 loads each, held in registers
  const bf16_t* qrow = Q + (size_t)(q0 + ln) * ND;
  const v16bf aq0 = load_a_frag(qrow, khalf);
  const v16bf aq1 = load_a_frag(qrow + 32, khalf);

  // --- scores = (Q K^T) / sqrt(D), tiles striped over 4 waves ---
  for (int t = wave; t < NL / 16; t += 4) {
    const int key0 = t * 16;
    const bf16_t* kp = K + (size_t)(key0 + ln) * ND + khalf * 16;
    const v16bf b0 = *(const v16bf*)(kp);
    const v16bf b1 = *(const v16bf*)(kp + 32);
    if (t + 4 < NL / 16)
      __builtin_prefetch(kp + 64 * ND, 0, 0);   // this wave's next K tile
    v8f s = {};
    s = __builtin_amdgcn_wmma_f32_16x16x32_bf16(
        false, aq0, false, b0, (short)0, s, false, false);
    s = __builtin_amdgcn_wmma_f32_16x16x32_bf16(
        false, aq1, false, b1, (short)0, s, false, false);
#pragma unroll
    for (int r = 0; r < 8; ++r)
      scores[(khalf * 8 + r) * SSTR + key0 + ln] = s[r] * 0.125f;
  }
  __syncthreads();

  // --- softmax over each of the 16 rows (8 threads per row) ---
  {
    const int row = tid >> 3;
    const int sub = tid & 7;
    float* srow = scores + row * SSTR;
    float pmax = -3.0e38f;
    for (int i = sub; i < NL; i += 8) pmax = fmaxf(pmax, srow[i]);
    red[row * 8 + sub] = pmax;
    __syncthreads();
    float rmax = red[row * 8];
#pragma unroll
    for (int j = 1; j < 8; ++j) rmax = fmaxf(rmax, red[row * 8 + j]);
    __syncthreads();
    float psum = 0.f;
    for (int i = sub; i < NL; i += 8) {
      const float e = __expf(srow[i] - rmax);
      srow[i] = e;
      psum += e;
    }
    red[row * 8 + sub] = psum;
    __syncthreads();
    float rsum = 0.f;
#pragma unroll
    for (int j = 0; j < 8; ++j) rsum += red[row * 8 + j];
    const float inv = 1.0f / rsum;
    for (int i = sub; i < NL; i += 8) srow[i] *= inv;
  }
  __syncthreads();

  // --- ctx = P @ V: A from LDS pairs (ds_load_b64 + v_perm), B = b128 loads
  v8f acc[4] = {{}, {}, {}, {}};
  for (int t = wave; t < NL / 32; t += 4) {
    const int kv0 = t * 32;
    v8u au;
#pragma unroll
    for (int j = 0; j < 8; ++j) {
      const int k = kv0 + khalf * 16 + j * 2;
      const float2 pv = *(const float2*)&scores[ln * SSTR + k];
      au[j] = pack_bf16(pv.x, pv.y);
    }
    const v16bf ap = __builtin_bit_cast(v16bf, au);
#pragma unroll
    for (int nt = 0; nt < 4; ++nt) {
      const bf16_t* vp = Vt + (size_t)(nt * 16 + ln) * NL + kv0 + khalf * 16;
      const v16bf bm = *(const v16bf*)vp;
      acc[nt] = __builtin_amdgcn_wmma_f32_16x16x32_bf16(
          false, ap, false, bm, (short)0, acc[nt], false, false);
    }
  }

  // cross-wave reduction of partial ctx, write bf16 ctx (B,L,C)
#pragma unroll
  for (int nt = 0; nt < 4; ++nt)
#pragma unroll
    for (int r = 0; r < 8; ++r)
      ctxRed[wave * 1024 + (khalf * 8 + r) * 64 + nt * 16 + ln] = acc[nt][r];
  __syncthreads();

  const int b = bh >> 3, h = bh & 7;
  for (int e = tid; e < 1024; e += 128) {
    const float sum = ctxRed[e] + ctxRed[1024 + e] +
                      ctxRed[2048 + e] + ctxRed[3072 + e];
    const int rw = e >> 6, col = e & 63;
    ctxbuf[((size_t)b * NL + q0 + rw) * NC + h * ND + col] = f2bf(sum);
  }
}

// ---------------------------------------------------------------------------
// K3: out = ctx @ Wo^T + bo + x, stored transposed to (B,C,L).
// grid=(L/16, C/64, B), 1 wave; 16x64 tile, 64 WMMAs.
// ---------------------------------------------------------------------------
__global__ __launch_bounds__(32) void out_kernel(
    const bf16_t* __restrict__ ctxbuf, const bf16_t* __restrict__ wob,
    const float* __restrict__ bo, const float* __restrict__ ppg,
    float* __restrict__ out)
{
  const int lid   = threadIdx.x & 31;
  const int ln    = lid & 15;
  const int khalf = lid >> 4;
  const int l0    = blockIdx.x * 16;
  const int c0    = blockIdx.y * 64;
  const int b     = blockIdx.z;

  const bf16_t* arow = ctxbuf + ((size_t)b * NL + l0 + ln) * NC;
  v8f acc[4] = {{}, {}, {}, {}};
#pragma unroll
  for (int kb = 0; kb < 16; ++kb) {
    const int kbase = kb * 32;
    const v16bf a = load_a_frag(arow + kbase, khalf);
#pragma unroll
    for (int nt = 0; nt < 4; ++nt) {
      const bf16_t* wrow =
          wob + (size_t)(c0 + nt * 16 + ln) * NC + kbase + khalf * 16;
      const v16bf bm = *(const v16bf*)wrow;
      acc[nt] = __builtin_amdgcn_wmma_f32_16x16x32_bf16(
          false, a, false, bm, (short)0, acc[nt], false, false);
    }
  }

#pragma unroll
  for (int nt = 0; nt < 4; ++nt) {
    const int c = c0 + nt * 16 + ln;
    const float bb = bo[c];
    const float* pr = ppg + ((size_t)b * NC + c) * NL + l0 + khalf * 8;
    float*       po = out + ((size_t)b * NC + c) * NL + l0 + khalf * 8;
#pragma unroll
    for (int r = 0; r < 8; ++r)
      po[r] = acc[nt][r] + bb + pr[r];  // residual with ppg^T, transposed out
  }
}

// ---------------------------------------------------------------------------
extern "C" void kernel_launch(void* const* d_in, const int* in_sizes, int n_in,
                              void* d_out, int out_size, void* d_ws, size_t ws_size,
                              hipStream_t stream) {
  (void)in_sizes; (void)n_in; (void)out_size; (void)ws_size;
  const float* ppg = (const float*)d_in[0];
  const float* ecg = (const float*)d_in[1];
  const float* Wq  = (const float*)d_in[2];
  const float* bq  = (const float*)d_in[3];
  const float* Wk  = (const float*)d_in[4];
  const float* bk  = (const float*)d_in[5];
  const float* Wv  = (const float*)d_in[6];
  const float* bv  = (const float*)d_in[7];
  const float* Wo  = (const float*)d_in[8];
  const float* bo  = (const float*)d_in[9];
  float* out = (float*)d_out;

  const size_t ACT = (size_t)NB * NL * NC;        // 4.19M bf16
  const size_t WSZ = (size_t)NC * NC;             // 256K bf16
  const size_t QKV = (size_t)NB * NH * NL * ND;   // 4.19M bf16
  bf16_t* xbf    = (bf16_t*)d_ws;
  bf16_t* ybf    = xbf + ACT;
  bf16_t* wqb    = ybf + ACT;
  bf16_t* wkb    = wqb + WSZ;
  bf16_t* wvb    = wkb + WSZ;
  bf16_t* wob    = wvb + WSZ;
  bf16_t* qbuf   = wob + WSZ;
  bf16_t* kbuf   = qbuf + QKV;
  bf16_t* vtbuf  = kbuf + QKV;
  bf16_t* ctxbuf = vtbuf + QKV;

  transpose_kernel<<<dim3(NL / 32, NC / 32, 2 * NB), 256, 0, stream>>>(
      ppg, ecg, xbf, ybf);
  wconv_kernel<<<dim3(2048), 256, 0, stream>>>(
      Wq, Wk, Wv, Wo, wqb, wkb, wvb, wob);
  qkv_kernel<<<dim3(NL / 16, NC / 64, 3 * NB), 32, 0, stream>>>(
      xbf, ybf, wqb, bq, wkb, bk, wvb, bv, qbuf, kbuf, vtbuf);
  const size_t smem = (size_t)(16 * SSTR + 16 * 8 + 4 * 16 * 64) * sizeof(float);
  attn_kernel<<<dim3(NL / 16, NB * NH), 128, smem, stream>>>(
      qbuf, kbuf, vtbuf, ctxbuf);
  out_kernel<<<dim3(NL / 16, NC / 64, NB), 32, 0, stream>>>(
      ctxbuf, wob, bo, ppg, out);
}
